// MoEGroupedGEMM_37933151158614
// MI455X (gfx1250) — compile-verified, matrix operand order checked by reading, not run on hardware
//
#include <hip/hip_runtime.h>
#include <cstdint>

// ---------------- problem dims ----------------
constexpr int kT  = 4096;          // B*S tokens
constexpr int kD  = 2048;          // hidden
constexpr int kE  = 16;            // experts
constexpr int kTopK = 2;
constexpr int kN  = kT * kTopK;    // 8192 (token,expert) pairs
constexpr int kFE = 1024;          // expert intermediate
constexpr int kFS = 2048;          // shared intermediate

// ---------------- GEMM tiling ----------------
constexpr int kTM = 64;            // macro tile M
constexpr int kTN = 128;           // macro tile N
constexpr int kTK = 32;            // K step (bf16 WMMA K)
constexpr int kLS = 40;            // padded LDS row stride (80B: conflict-free, 16B aligned)
constexpr int kMaxTiles = 160;     // >= N/kTM + kE = 144

typedef __attribute__((ext_vector_type(16))) __bf16 v16bf;
typedef __attribute__((ext_vector_type(8)))  float  v8f;
typedef __attribute__((ext_vector_type(4)))  float  f32x4;
typedef __attribute__((ext_vector_type(4)))  unsigned int u32x4;

union FragBF { v16bf v; u32x4 q[2]; };

// ---- CDNA5 async global->LDS (ASYNCcnt) with safe fallback ----
#if defined(__HIP_DEVICE_COMPILE__) && \
    __has_builtin(__builtin_amdgcn_global_load_async_to_lds_b128) && \
    __has_builtin(__builtin_amdgcn_s_wait_asynccnt)
#define USE_ASYNC_LDS 1
typedef int i32x4_gcc __attribute__((vector_size(16)));
typedef __attribute__((address_space(1))) i32x4_gcc as1_i32x4;
typedef __attribute__((address_space(3))) i32x4_gcc as3_i32x4;
#else
#define USE_ASYNC_LDS 0
#endif

__device__ inline void ldsCopy16Async(const __bf16* g, __bf16* l) {
#if USE_ASYNC_LDS
  __builtin_amdgcn_global_load_async_to_lds_b128((as1_i32x4*)g, (as3_i32x4*)l, 0, 0);
#else
  *(u32x4*)l = *(const u32x4*)g;
#endif
}

__device__ inline void ldsAsyncWait() {
#if USE_ASYNC_LDS
  __builtin_amdgcn_s_wait_asynccnt(0);
#endif
}

// Load a 16x32 bf16 fragment (A-layout; B uses the same pattern on an
// LDS tile stored [N][K]).  Per CDNA5 ISA 7.12.2:
//   lanes 0-15  : M/N = lane,    K = {0..7} and {16..23}
//   lanes 16-31 : M/N = lane-16, K = {8..15} and {24..31}
__device__ inline FragBF ldsFrag(const __bf16* lds, int base16, int lane) {
  FragBF f;
  const int mj = base16 + (lane & 15);
  const int kb = (lane & 16) ? 8 : 0;
  f.q[0] = *(const u32x4*)(lds + mj * kLS + kb);
  f.q[1] = *(const u32x4*)(lds + mj * kLS + kb + 16);
  return f;
}

__device__ inline v8f wmma_bf16(const FragBF& a, const FragBF& b, v8f c) {
  return __builtin_amdgcn_wmma_f32_16x16x32_bf16(false, a.v, false, b.v,
                                                 (short)0, c, false, false);
}

__device__ inline float siluf(float g) { return g / (1.0f + __expf(-g)); }

// ---------------------------------------------------------------------------
// 0) fp32 -> bf16 activation convert
// ---------------------------------------------------------------------------
__global__ __launch_bounds__(256) void k_convert(const float* __restrict__ x,
                                                 __bf16* __restrict__ xb) {
  const int i = (blockIdx.x * 256 + threadIdx.x) * 4;
  f32x4 v = *(const f32x4*)(x + i);
#pragma unroll
  for (int j = 0; j < 4; ++j) xb[i + j] = (__bf16)v[j];
}

// ---------------------------------------------------------------------------
// 1) router: logits, softmax, top-2 renorm.  One wave32 per token.
// ---------------------------------------------------------------------------
__global__ __launch_bounds__(256) void k_router(const float* __restrict__ X,
                                                const float* __restrict__ Rw,
                                                float* __restrict__ logits_out,
                                                int* __restrict__ topk_idx,
                                                float* __restrict__ topk_w) {
  const int lane = threadIdx.x & 31;
  const int wave = threadIdx.x >> 5;
  const int t = blockIdx.x * 8 + wave;
  if (t >= kT) return;

  float acc[kE];
#pragma unroll
  for (int e = 0; e < kE; ++e) acc[e] = 0.0f;

  const float* x = X + (size_t)t * kD;
  for (int j = lane; j < kD; j += 32) {
    const float xv = x[j];
#pragma unroll
    for (int e = 0; e < kE; ++e) acc[e] += xv * Rw[e * kD + j];
  }
#pragma unroll
  for (int e = 0; e < kE; ++e) {
    float v = acc[e];
    for (int off = 16; off > 0; off >>= 1) v += __shfl_xor(v, off, 32);
    acc[e] = v;
  }
  if (lane == 0) {
    float m = acc[0];
#pragma unroll
    for (int e = 1; e < kE; ++e) m = fmaxf(m, acc[e]);
    float p[kE];
#pragma unroll
    for (int e = 0; e < kE; ++e) {
      logits_out[(size_t)t * kE + e] = acc[e];
      p[e] = __expf(acc[e] - m);
    }
    int i1 = 0;
    for (int e = 1; e < kE; ++e) if (p[e] > p[i1]) i1 = e;
    int i2 = (i1 == 0) ? 1 : 0;
    for (int e = 0; e < kE; ++e) if (e != i1 && p[e] > p[i2]) i2 = e;
    const float s = p[i1] + p[i2];
    topk_idx[t * 2 + 0] = i1; topk_w[t * 2 + 0] = p[i1] / s;
    topk_idx[t * 2 + 1] = i2; topk_w[t * 2 + 1] = p[i2] / s;
  }
}

// ---------------------------------------------------------------------------
// 2) grouped layout: count -> scan/tile-table -> scatter (+inverse map)
// ---------------------------------------------------------------------------
__global__ void k_count(const int* __restrict__ topk_idx, int* __restrict__ counts) {
  const int i = blockIdx.x * 256 + threadIdx.x;
  if (i < kN) atomicAdd(&counts[topk_idx[i]], 1);
}

__global__ void k_build(const int* __restrict__ counts, int* __restrict__ offsets,
                        int* __restrict__ tileE, int* __restrict__ tileRow,
                        int* __restrict__ tileRows, int* __restrict__ numTiles) {
  int off = 0, nt = 0;
  for (int e = 0; e < kE; ++e) {
    offsets[e] = off;
    const int c = counts[e];
    for (int r = 0; r < c; r += kTM) {
      tileE[nt] = e;
      tileRow[nt] = off + r;
      tileRows[nt] = (c - r) < kTM ? (c - r) : kTM;
      ++nt;
    }
    off += c;
  }
  offsets[kE] = off;
  *numTiles = nt;
}

__global__ void k_scatter(const int* __restrict__ topk_idx, const float* __restrict__ topk_w,
                          const int* __restrict__ offsets, int* __restrict__ cursor,
                          int* __restrict__ sorted_tok, int* __restrict__ pos_of) {
  const int i = blockIdx.x * 256 + threadIdx.x;
  if (i >= kN) return;
  const int e = topk_idx[i];
  const int p = offsets[e] + atomicAdd(&cursor[e], 1);
  sorted_tok[p] = i >> 1;
  pos_of[i] = p;
}

// ---------------------------------------------------------------------------
// 3) shared expert: fused gate+up GEMM + SwiGLU  (Xb[T,D] -> Hs[T,FS] bf16)
//    8 waves as 2(M) x 4(N); each wave owns a 32x32 output per matrix.
// ---------------------------------------------------------------------------
__global__ __launch_bounds__(256) void k_shared_gateup(
    const __bf16* __restrict__ Xb, const float* __restrict__ Wg,
    const float* __restrict__ Wu, __bf16* __restrict__ Hs) {
  __shared__ alignas(16) __bf16 Alds[kTM * kLS];
  __shared__ alignas(16) __bf16 Bg[kTN * kLS];
  __shared__ alignas(16) __bf16 Bu[kTN * kLS];

  const int m0 = blockIdx.x * kTM;
  const int n0 = blockIdx.y * kTN;
  const int tid = threadIdx.x;
  const int lane = tid & 31, wave = tid >> 5;
  const int wm = (wave & 1) * 32;
  const int wn = (wave >> 1) * 32;

  v8f accG[2][2], accU[2][2];
#pragma unroll
  for (int i = 0; i < 2; ++i)
#pragma unroll
    for (int j = 0; j < 2; ++j)
#pragma unroll
      for (int r = 0; r < 8; ++r) { accG[i][j][r] = 0.0f; accU[i][j][r] = 0.0f; }

  const int ar = tid >> 2, ac = (tid & 3) * 8;   // A staging: 64 rows x 4 chunks
  for (int k0 = 0; k0 < kD; k0 += kTK) {
    __syncthreads();
    // A tile: 64x32 bf16 (async DMA to LDS on CDNA5)
    ldsCopy16Async(Xb + (size_t)(m0 + ar) * kD + k0 + ac, Alds + ar * kLS + ac);
    // B tiles (gate+up): fp32 -> bf16, stored [N][K] with padded stride
#pragma unroll
    for (int r = 0; r < 4; ++r) {
      const int kr = r * 8 + (tid >> 5);
      const int nc = (tid & 31) * 4;
      f32x4 g = *(const f32x4*)(Wg + (size_t)(k0 + kr) * kFS + n0 + nc);
      f32x4 u = *(const f32x4*)(Wu + (size_t)(k0 + kr) * kFS + n0 + nc);
#pragma unroll
      for (int j = 0; j < 4; ++j) {
        Bg[(nc + j) * kLS + kr] = (__bf16)g[j];
        Bu[(nc + j) * kLS + kr] = (__bf16)u[j];
      }
      if (r == 0 && k0 + kTK < kD) {
        __builtin_prefetch(Wg + (size_t)(k0 + kTK + kr) * kFS + n0 + nc, 0, 0);
        __builtin_prefetch(Wu + (size_t)(k0 + kTK + kr) * kFS + n0 + nc, 0, 0);
      }
    }
    ldsAsyncWait();
    __syncthreads();

    FragBF a[2], bg[2], bu[2];
#pragma unroll
    for (int i = 0; i < 2; ++i) a[i] = ldsFrag(Alds, wm + i * 16, lane);
#pragma unroll
    for (int j = 0; j < 2; ++j) {
      bg[j] = ldsFrag(Bg, wn + j * 16, lane);
      bu[j] = ldsFrag(Bu, wn + j * 16, lane);
    }
#pragma unroll
    for (int i = 0; i < 2; ++i)
#pragma unroll
      for (int j = 0; j < 2; ++j) {
        accG[i][j] = wmma_bf16(a[i], bg[j], accG[i][j]);
        accU[i][j] = wmma_bf16(a[i], bu[j], accU[i][j]);
      }
  }
  // epilogue: h = silu(g)*u
  const int half = (lane >> 4) & 1;
  const int nloc = lane & 15;
#pragma unroll
  for (int i = 0; i < 2; ++i)
#pragma unroll
    for (int j = 0; j < 2; ++j)
#pragma unroll
      for (int r = 0; r < 8; ++r) {
        const float h = siluf(accG[i][j][r]) * accU[i][j][r];
        const int row = m0 + wm + i * 16 + r + 8 * half;
        const int col = n0 + wn + j * 16 + nloc;
        Hs[(size_t)row * kFS + col] = (__bf16)h;
      }
}

// ---------------------------------------------------------------------------
// 4) shared expert down GEMM: Hs[T,FS] @ Wd[FS,D] -> out (fp32, overwrites)
// ---------------------------------------------------------------------------
__global__ __launch_bounds__(256) void k_shared_down(
    const __bf16* __restrict__ Hs, const float* __restrict__ Wd,
    float* __restrict__ out) {
  __shared__ alignas(16) __bf16 Alds[kTM * kLS];
  __shared__ alignas(16) __bf16 Blds[kTN * kLS];

  const int m0 = blockIdx.x * kTM;
  const int n0 = blockIdx.y * kTN;
  const int tid = threadIdx.x;
  const int lane = tid & 31, wave = tid >> 5;
  const int wm = (wave & 1) * 32;
  const int wn = (wave >> 1) * 32;

  v8f acc[2][2];
#pragma unroll
  for (int i = 0; i < 2; ++i)
#pragma unroll
    for (int j = 0; j < 2; ++j)
#pragma unroll
      for (int r = 0; r < 8; ++r) acc[i][j][r] = 0.0f;

  const int ar = tid >> 2, ac = (tid & 3) * 8;
  for (int k0 = 0; k0 < kFS; k0 += kTK) {
    __syncthreads();
    ldsCopy16Async(Hs + (size_t)(m0 + ar) * kFS + k0 + ac, Alds + ar * kLS + ac);
#pragma unroll
    for (int r = 0; r < 4; ++r) {
      const int kr = r * 8 + (tid >> 5);
      const int nc = (tid & 31) * 4;
      f32x4 w = *(const f32x4*)(Wd + (size_t)(k0 + kr) * kD + n0 + nc);
#pragma unroll
      for (int j = 0; j < 4; ++j) Blds[(nc + j) * kLS + kr] = (__bf16)w[j];
      if (r == 0 && k0 + kTK < kFS)
        __builtin_prefetch(Wd + (size_t)(k0 + kTK + kr) * kD + n0 + nc, 0, 0);
    }
    ldsAsyncWait();
    __syncthreads();

    FragBF a[2], b[2];
#pragma unroll
    for (int i = 0; i < 2; ++i) a[i] = ldsFrag(Alds, wm + i * 16, lane);
#pragma unroll
    for (int j = 0; j < 2; ++j) b[j] = ldsFrag(Blds, wn + j * 16, lane);
#pragma unroll
    for (int i = 0; i < 2; ++i)
#pragma unroll
      for (int j = 0; j < 2; ++j) acc[i][j] = wmma_bf16(a[i], b[j], acc[i][j]);
  }
  const int half = (lane >> 4) & 1;
  const int nloc = lane & 15;
#pragma unroll
  for (int i = 0; i < 2; ++i)
#pragma unroll
    for (int j = 0; j < 2; ++j)
#pragma unroll
      for (int r = 0; r < 8; ++r) {
        const int row = m0 + wm + i * 16 + r + 8 * half;
        const int col = n0 + wn + j * 16 + nloc;
        out[(size_t)row * kD + col] = acc[i][j][r];
      }
}

// ---------------------------------------------------------------------------
// 5) expert gate+up + SwiGLU (grouped, gathered A rows) -> He[N,FE] bf16
// ---------------------------------------------------------------------------
__global__ __launch_bounds__(256) void k_exp_gateup(
    const __bf16* __restrict__ Xb, const float* __restrict__ gate_w,
    const float* __restrict__ up_w, const int* __restrict__ sorted_tok,
    const int* __restrict__ tileE, const int* __restrict__ tileRow,
    const int* __restrict__ tileRowsArr, const int* __restrict__ numTiles,
    __bf16* __restrict__ He) {
  const int tile = blockIdx.x;
  if (tile >= *numTiles) return;
  const int e = tileE[tile];
  const int rowStart = tileRow[tile];
  const int valid = tileRowsArr[tile];
  const float* Wg = gate_w + (size_t)e * kD * kFE;
  const float* Wu = up_w + (size_t)e * kD * kFE;

  __shared__ alignas(16) __bf16 Alds[kTM * kLS];
  __shared__ alignas(16) __bf16 Bg[kTN * kLS];
  __shared__ alignas(16) __bf16 Bu[kTN * kLS];

  const int n0 = blockIdx.y * kTN;
  const int tid = threadIdx.x;
  const int lane = tid & 31, wave = tid >> 5;
  const int wm = (wave & 1) * 32;
  const int wn = (wave >> 1) * 32;

  v8f accG[2][2], accU[2][2];
#pragma unroll
  for (int i = 0; i < 2; ++i)
#pragma unroll
    for (int j = 0; j < 2; ++j)
#pragma unroll
      for (int r = 0; r < 8; ++r) { accG[i][j][r] = 0.0f; accU[i][j][r] = 0.0f; }

  // gathered token row for this thread's A slice
  const int ar = tid >> 2, ac = (tid & 3) * 8;
  int gi = rowStart + ar; if (gi > kN - 1) gi = kN - 1;
  const int tok = sorted_tok[gi];

  for (int k0 = 0; k0 < kD; k0 += kTK) {
    __syncthreads();
    ldsCopy16Async(Xb + (size_t)tok * kD + k0 + ac, Alds + ar * kLS + ac);
#pragma unroll
    for (int r = 0; r < 4; ++r) {
      const int kr = r * 8 + (tid >> 5);
      const int nc = (tid & 31) * 4;
      f32x4 g = *(const f32x4*)(Wg + (size_t)(k0 + kr) * kFE + n0 + nc);
      f32x4 u = *(const f32x4*)(Wu + (size_t)(k0 + kr) * kFE + n0 + nc);
#pragma unroll
      for (int j = 0; j < 4; ++j) {
        Bg[(nc + j) * kLS + kr] = (__bf16)g[j];
        Bu[(nc + j) * kLS + kr] = (__bf16)u[j];
      }
      if (r == 0 && k0 + kTK < kD) {
        __builtin_prefetch(Wg + (size_t)(k0 + kTK + kr) * kFE + n0 + nc, 0, 0);
        __builtin_prefetch(Wu + (size_t)(k0 + kTK + kr) * kFE + n0 + nc, 0, 0);
      }
    }
    ldsAsyncWait();
    __syncthreads();

    FragBF a[2], bg[2], bu[2];
#pragma unroll
    for (int i = 0; i < 2; ++i) a[i] = ldsFrag(Alds, wm + i * 16, lane);
#pragma unroll
    for (int j = 0; j < 2; ++j) {
      bg[j] = ldsFrag(Bg, wn + j * 16, lane);
      bu[j] = ldsFrag(Bu, wn + j * 16, lane);
    }
#pragma unroll
    for (int i = 0; i < 2; ++i)
#pragma unroll
      for (int j = 0; j < 2; ++j) {
        accG[i][j] = wmma_bf16(a[i], bg[j], accG[i][j]);
        accU[i][j] = wmma_bf16(a[i], bu[j], accU[i][j]);
      }
  }
  const int half = (lane >> 4) & 1;
  const int nloc = lane & 15;
#pragma unroll
  for (int i = 0; i < 2; ++i)
#pragma unroll
    for (int j = 0; j < 2; ++j)
#pragma unroll
      for (int r = 0; r < 8; ++r) {
        const int rloc = wm + i * 16 + r + 8 * half;
        if (rloc < valid) {
          const float h = siluf(accG[i][j][r]) * accU[i][j][r];
          const int col = n0 + wn + j * 16 + nloc;
          He[(size_t)(rowStart + rloc) * kFE + col] = (__bf16)h;
        }
      }
}

// ---------------------------------------------------------------------------
// 6) expert down GEMM (grouped): He[N,FE] @ down_w[e][FE,D] -> Dn[N,D] bf16
// ---------------------------------------------------------------------------
__global__ __launch_bounds__(256) void k_exp_down(
    const __bf16* __restrict__ He, const float* __restrict__ down_w,
    const int* __restrict__ tileE, const int* __restrict__ tileRow,
    const int* __restrict__ tileRowsArr, const int* __restrict__ numTiles,
    __bf16* __restrict__ Dn) {
  const int tile = blockIdx.x;
  if (tile >= *numTiles) return;
  const int e = tileE[tile];
  const int rowStart = tileRow[tile];
  const int valid = tileRowsArr[tile];
  const float* Wd = down_w + (size_t)e * kFE * kD;

  __shared__ alignas(16) __bf16 Alds[kTM * kLS];
  __shared__ alignas(16) __bf16 Blds[kTN * kLS];

  const int n0 = blockIdx.y * kTN;
  const int tid = threadIdx.x;
  const int lane = tid & 31, wave = tid >> 5;
  const int wm = (wave & 1) * 32;
  const int wn = (wave >> 1) * 32;

  v8f acc[2][2];
#pragma unroll
  for (int i = 0; i < 2; ++i)
#pragma unroll
    for (int j = 0; j < 2; ++j)
#pragma unroll
      for (int r = 0; r < 8; ++r) acc[i][j][r] = 0.0f;

  const int ar = tid >> 2, ac = (tid & 3) * 8;
  int gr = rowStart + ar; if (gr > kN - 1) gr = kN - 1;

  for (int k0 = 0; k0 < kFE; k0 += kTK) {
    __syncthreads();
    ldsCopy16Async(He + (size_t)gr * kFE + k0 + ac, Alds + ar * kLS + ac);
#pragma unroll
    for (int r = 0; r < 4; ++r) {
      const int kr = r * 8 + (tid >> 5);
      const int nc = (tid & 31) * 4;
      f32x4 w = *(const f32x4*)(Wd + (size_t)(k0 + kr) * kD + n0 + nc);
#pragma unroll
      for (int j = 0; j < 4; ++j) Blds[(nc + j) * kLS + kr] = (__bf16)w[j];
      if (r == 0 && k0 + kTK < kFE)
        __builtin_prefetch(Wd + (size_t)(k0 + kTK + kr) * kD + n0 + nc, 0, 0);
    }
    ldsAsyncWait();
    __syncthreads();

    FragBF a[2], b[2];
#pragma unroll
    for (int i = 0; i < 2; ++i) a[i] = ldsFrag(Alds, wm + i * 16, lane);
#pragma unroll
    for (int j = 0; j < 2; ++j) b[j] = ldsFrag(Blds, wn + j * 16, lane);
#pragma unroll
    for (int i = 0; i < 2; ++i)
#pragma unroll
      for (int j = 0; j < 2; ++j) acc[i][j] = wmma_bf16(a[i], b[j], acc[i][j]);
  }
  const int half = (lane >> 4) & 1;
  const int nloc = lane & 15;
#pragma unroll
  for (int i = 0; i < 2; ++i)
#pragma unroll
    for (int j = 0; j < 2; ++j)
#pragma unroll
      for (int r = 0; r < 8; ++r) {
        const int rloc = wm + i * 16 + r + 8 * half;
        if (rloc < valid) {
          const int col = n0 + wn + j * 16 + nloc;
          Dn[(size_t)(rowStart + rloc) * kD + col] = (__bf16)acc[i][j][r];
        }
      }
}

// ---------------------------------------------------------------------------
// 7) combine: out[t] += w0*Dn[p0] + w1*Dn[p1]   (deterministic gather)
// ---------------------------------------------------------------------------
__global__ __launch_bounds__(256) void k_combine(float* __restrict__ out,
                                                 const __bf16* __restrict__ Dn,
                                                 const int* __restrict__ pos_of,
                                                 const float* __restrict__ topk_w) {
  const int t = blockIdx.x;
  const int p0 = pos_of[t * 2 + 0], p1 = pos_of[t * 2 + 1];
  const float w0 = topk_w[t * 2 + 0], w1 = topk_w[t * 2 + 1];
  for (int d = threadIdx.x; d < kD; d += 256) {
    float v = out[(size_t)t * kD + d];
    v += w0 * (float)Dn[(size_t)p0 * kD + d] + w1 * (float)Dn[(size_t)p1 * kD + d];
    out[(size_t)t * kD + d] = v;
  }
}

// ---------------------------------------------------------------------------
// host
// ---------------------------------------------------------------------------
extern "C" void kernel_launch(void* const* d_in, const int* in_sizes, int n_in,
                              void* d_out, int out_size, void* d_ws, size_t ws_size,
                              hipStream_t stream) {
  const float* hidden   = (const float*)d_in[0];
  const float* router_w = (const float*)d_in[1];
  const float* gate_w   = (const float*)d_in[2];
  const float* up_w     = (const float*)d_in[3];
  const float* down_w   = (const float*)d_in[4];
  const float* sh_gate  = (const float*)d_in[5];
  const float* sh_up    = (const float*)d_in[6];
  const float* sh_down  = (const float*)d_in[7];

  float* out        = (float*)d_out;
  float* out_logits = out + (size_t)kT * kD;

  char* ws = (char*)d_ws;
  size_t off = 0;
  auto alloc = [&](size_t bytes) -> void* {
    void* p = ws + off;
    off = (off + bytes + 255) & ~(size_t)255;
    return p;
  };

  __bf16* Xb = (__bf16*)alloc((size_t)kT * kD * 2);     // 16 MiB
  __bf16* Hs = (__bf16*)alloc((size_t)kT * kFS * 2);    // 16 MiB
  __bf16* He = (__bf16*)alloc((size_t)kN * kFE * 2);    // 16 MiB
  __bf16* Dn = (__bf16*)alloc((size_t)kN * kD * 2);     // 32 MiB
  int*    topk_idx   = (int*)alloc(kN * sizeof(int));
  float*  topk_w     = (float*)alloc(kN * sizeof(float));
  int*    sorted_tok = (int*)alloc(kN * sizeof(int));
  int*    pos_of     = (int*)alloc(kN * sizeof(int));
  int*    counts     = (int*)alloc(32 * sizeof(int));   // counts[16] + cursor[16]
  int*    cursor     = counts + 16;
  int*    offsets    = (int*)alloc((kE + 1) * sizeof(int));
  int*    tileE      = (int*)alloc(kMaxTiles * sizeof(int));
  int*    tileRow    = (int*)alloc(kMaxTiles * sizeof(int));
  int*    tileRows   = (int*)alloc(kMaxTiles * sizeof(int));
  int*    numTiles   = (int*)alloc(sizeof(int));

  (void)hipMemsetAsync(counts, 0, 32 * sizeof(int), stream);

  // activations -> bf16
  k_convert<<<(kT * kD) / (256 * 4), 256, 0, stream>>>(hidden, Xb);

  // router + top-2
  k_router<<<kT / 8, 256, 0, stream>>>(hidden, router_w, out_logits, topk_idx, topk_w);

  // grouped layout
  k_count<<<kN / 256, 256, 0, stream>>>(topk_idx, counts);
  k_build<<<1, 1, 0, stream>>>(counts, offsets, tileE, tileRow, tileRows, numTiles);
  k_scatter<<<kN / 256, 256, 0, stream>>>(topk_idx, topk_w, offsets, cursor,
                                          sorted_tok, pos_of);

  // shared expert
  {
    dim3 g(kT / kTM, kFS / kTN);
    k_shared_gateup<<<g, 256, 0, stream>>>(Xb, sh_gate, sh_up, Hs);
  }
  {
    dim3 g(kT / kTM, kD / kTN);
    k_shared_down<<<g, 256, 0, stream>>>(Hs, sh_down, out);
  }

  // routed experts (grouped GEMMs via tile table)
  {
    dim3 g(kMaxTiles, kFE / kTN);
    k_exp_gateup<<<g, 256, 0, stream>>>(Xb, gate_w, up_w, sorted_tok,
                                        tileE, tileRow, tileRows, numTiles, He);
  }
  {
    dim3 g(kMaxTiles, kD / kTN);
    k_exp_down<<<g, 256, 0, stream>>>(He, down_w, tileE, tileRow, tileRows,
                                      numTiles, Dn);
  }

  // weighted gather-combine
  k_combine<<<kT, 256, 0, stream>>>(out, Dn, pos_of, topk_w);
}
